// AIGGenerator_68951404970178
// MI455X (gfx1250) — matrix operand order repaired and checked by reading, not compiled, not run on hardware
//
#include <hip/hip_runtime.h>
#include <hip/hip_bf16.h>

// ---------------------------------------------------------------------------
// AIG generator pipeline for MI455X (gfx1250, wave32, WMMA).
//
// Dominant cost: scores = Ht @ Hs^T (8192x8192x128 ~ 17.2 GFLOP). We fuse the
// GEMM with depth masking + per-row top-2 so the 256MB score matrix never
// touches memory. All dense GEMMs use v_wmma_f32_16x16x32_f16 (f16 inputs,
// f32 accumulate). Inputs to the WMMA GEMMs (4MB f32 -> 2MB f16 each) fit in
// L2 (192MB); nothing here is HBM-bound at 23.3 TB/s, so the kernel set is
// engineered around keeping the WMMA pipe busy and avoiding the NxN blowup.
// ---------------------------------------------------------------------------

#define NNODES 8192
#define NEDGES 131072
#define HDIM   128
#define ZDIM   128
#define NEGV   (-1e30f)

typedef __attribute__((ext_vector_type(16))) _Float16 v16h;
typedef __attribute__((ext_vector_type(8)))  _Float16 v8h;
typedef __attribute__((ext_vector_type(8)))  float    v8f;

// ---------------- WMMA fragment loaders (16-bit, wave32 layouts) -----------
// A matrix 16x32 (MxK): lane l holds row (l&15); halves
// K = k0 + (l>>4)*16 .. +15 (two 16B loads).
static __device__ inline v16h load_a_frag(const _Float16* A, int lda,
                                          int row_base, int k0, int lane) {
    int r  = lane & 15;
    int hi = lane >> 4;
    const _Float16* p = A + (size_t)(row_base + r) * lda + k0 + hi * 16;
    v8h lo = *(const v8h*)p;
    v8h hi8 = *(const v8h*)(p + 8);
    return __builtin_shufflevector(lo, hi8,
        0, 1, 2, 3, 4, 5, 6, 7, 8, 9, 10, 11, 12, 13, 14, 15);
}

// B matrix 32x16 (KxN): lane l holds column (l&15); halves
// K = k0 + (l>>4)*16 .. +15 (one contiguous 32B load). Column c of B is
// row (col_base + c) of a row-major [Nout,K] matrix (NT GEMM).
static __device__ inline v16h load_b_frag(const _Float16* B, int ldb,
                                          int col_base, int k0, int lane) {
    int c    = lane & 15;
    int koff = (lane >> 4) * 16;
    const _Float16* p = B + (size_t)(col_base + c) * ldb + k0 + koff;
    return *(const v16h*)p;
}

// ---------------- small utility kernels ------------------------------------
__global__ void k_fill_deg(const int* dst, float* deg, int e) {
    int i = blockIdx.x * blockDim.x + threadIdx.x;
    if (i < e) atomicAdd(&deg[dst[i]], 1.0f);
}

__global__ void k_node_prep(const int* depth, float* deg, float* rsq,
                            int* lim, int n) {
    int i = blockIdx.x * blockDim.x + threadIdx.x;
    if (i >= n) return;
    float d = deg[i] + 1.0f;          // self loop
    deg[i] = d;
    rsq[i] = rsqrtf(d);
    // depth is sorted ascending: lim = #u with depth[u] <= depth[i] + 1
    int thr = depth[i] + 1;           // -1 + DEPTH_PERTURB(2)
    int lo = 0, hi = n;
    while (lo < hi) {
        int mid = (lo + hi) >> 1;
        if (depth[mid] <= thr) lo = mid + 1; else hi = mid;
    }
    lim[i] = lo;
}

__global__ void k_conv1(const float* x, const float* w, float* xw, int n) {
    int t = blockIdx.x * blockDim.x + threadIdx.x;
    if (t >= n * HDIM) return;
    int i = t >> 7, o = t & 127;
    xw[t] = x[i * 2] * w[o * 2] + x[i * 2 + 1] * w[o * 2 + 1];
}

__global__ void k_scatter(const int* src, const int* dst, const float* rsq,
                          const float* xw, float* agg, int e) {
    int t = blockIdx.x * blockDim.x + threadIdx.x;
    if (t >= e * HDIM) return;
    int ei = t >> 7, f = t & 127;
    int s = src[ei], d = dst[ei];
    atomicAdd(&agg[(size_t)d * HDIM + f], rsq[s] * rsq[d] * xw[(size_t)s * HDIM + f]);
}

__global__ void k_gcn_fin(const float* agg, const float* xw, const float* b,
                          const float* deg, float* h, int n) {
    int t = blockIdx.x * blockDim.x + threadIdx.x;
    if (t >= n * HDIM) return;
    int i = t >> 7, f = t & 127;
    float v = agg[t] + xw[t] / deg[i] + b[f];
    h[t] = fmaxf(v, 0.0f);
}

__global__ void k_f32_to_f16(const float* in, _Float16* out, int n) {
    int i = blockIdx.x * blockDim.x + threadIdx.x;
    if (i < n) out[i] = (_Float16)in[i];
}

__global__ void k_build_hz16(const float* h, const float* z, _Float16* hz,
                             int n) {
    int t = blockIdx.x * blockDim.x + threadIdx.x;
    if (t >= n * 256) return;
    int i = t >> 8, c = t & 255;
    hz[t] = (_Float16)(c < HDIM ? h[(size_t)i * HDIM + c] : z[c - HDIM]);
}

// ---------------- generic WMMA NT GEMM: C[M,128] = A[M,K] * W[128,K]^T -----
// grid = M/16 blocks, 256 threads (8 waves, one 16x16 column tile each).
__global__ __launch_bounds__(256)
void k_gemm_nt(const _Float16* __restrict__ A, const _Float16* __restrict__ W,
               float* __restrict__ C, const float* __restrict__ bias,
               int K, int relu) {
    int wave = threadIdx.x >> 5;
    int lane = threadIdx.x & 31;
    int mbase = blockIdx.x * 16;
    int nbase = wave * 16;
    v8f acc = {};
    for (int k0 = 0; k0 < K; k0 += 32) {
        v16h a = load_a_frag(A, K, mbase, k0, lane);
        v16h b = load_b_frag(W, K, nbase, k0, lane);
        acc = __builtin_amdgcn_wmma_f32_16x16x32_f16(
            false, a, false, b, (short)0, acc, false, false);
    }
    int col = nbase + (lane & 15);
    int hi  = lane >> 4;
    float bv = bias ? bias[col] : 0.0f;
#pragma unroll
    for (int r = 0; r < 8; ++r) {
        float v = acc[r] + bv;
        if (relu) v = fmaxf(v, 0.0f);
        C[(size_t)(mbase + r + hi * 8) * HDIM + col] = v;
    }
}

// ---------------- fused scores GEMM + depth mask + top-2 -------------------
static __device__ inline void top2_ins(float v, int i, float& v1, int& i1,
                                       float& v2, int& i2) {
    if (v > v1) { v2 = v1; i2 = i1; v1 = v; i1 = i; }
    else if (v > v2) { v2 = v; i2 = i; }
}

// grid = N/16 blocks (one 16-target-row stripe each), 256 threads (8 waves).
// Wave w streams u-tiles w, w+8, ... ; per-lane running top-2; xor-butterfly
// merge over 16 lanes; LDS merge across waves.
__global__ __launch_bounds__(256)
void k_scores_top2(const _Float16* __restrict__ Ht,
                   const _Float16* __restrict__ Hs,
                   const int* __restrict__ lim,
                   float* __restrict__ out, int* __restrict__ topidx, int n) {
    int wave = threadIdx.x >> 5;
    int lane = threadIdx.x & 31;
    int vbase = blockIdx.x * 16;
    int hi = lane >> 4;
    int cl = lane & 15;

    // A fragments for the full K=128, loaded once per wave.
    v16h afrag[4];
#pragma unroll
    for (int kk = 0; kk < 4; ++kk)
        afrag[kk] = load_a_frag(Ht, HDIM, vbase, kk * 32, lane);

    int mylim[8];
#pragma unroll
    for (int r = 0; r < 8; ++r) mylim[r] = lim[vbase + hi * 8 + r];

    float v1[8], v2[8]; int i1[8], i2[8];
#pragma unroll
    for (int r = 0; r < 8; ++r) { v1[r] = NEGV; v2[r] = NEGV; i1[r] = 0; i2[r] = 0; }

    int ntile = n >> 4;
    for (int ut = wave; ut < ntile; ut += 8) {
        int ubase = ut * 16;
        v8f acc = {};
#pragma unroll
        for (int kk = 0; kk < 4; ++kk) {
            v16h b = load_b_frag(Hs, HDIM, ubase, kk * 32, lane);
            acc = __builtin_amdgcn_wmma_f32_16x16x32_f16(
                false, afrag[kk], false, b, (short)0, acc, false, false);
        }
        int u = ubase + cl;
#pragma unroll
        for (int r = 0; r < 8; ++r) {
            float val = (u < mylim[r]) ? acc[r] : NEGV;
            top2_ins(val, u, v1[r], i1[r], v2[r], i2[r]);
        }
    }

    // merge top-2 sets across the 16 lanes of each half (columns).
#pragma unroll
    for (int m = 1; m < 16; m <<= 1) {
#pragma unroll
        for (int r = 0; r < 8; ++r) {
            float ov1 = __shfl_xor(v1[r], m, 32);
            int   oi1 = __shfl_xor(i1[r], m, 32);
            float ov2 = __shfl_xor(v2[r], m, 32);
            int   oi2 = __shfl_xor(i2[r], m, 32);
            top2_ins(ov1, oi1, v1[r], i1[r], v2[r], i2[r]);
            top2_ins(ov2, oi2, v1[r], i1[r], v2[r], i2[r]);
        }
    }

    __shared__ float sv1[8][16], sv2[8][16];
    __shared__ int   si1[8][16], si2[8][16];
    if (cl == 0) {
#pragma unroll
        for (int r = 0; r < 8; ++r) {
            int row = hi * 8 + r;
            sv1[wave][row] = v1[r]; si1[wave][row] = i1[r];
            sv2[wave][row] = v2[r]; si2[wave][row] = i2[r];
        }
    }
    __syncthreads();

    if (threadIdx.x < 16) {
        int row = threadIdx.x;
        float b1 = NEGV, b2 = NEGV; int j1 = 0, j2 = 0;
#pragma unroll
        for (int w = 0; w < 8; ++w) {
            top2_ins(sv1[w][row], si1[w][row], b1, j1, b2, j2);
            top2_ins(sv2[w][row], si2[w][row], b1, j1, b2, j2);
        }
        int v = vbase + row;
        int NO = n * 2;
        out[v * 2 + 0] = b1;                    // top_vals
        out[v * 2 + 1] = b2;
        out[2 * NO + v * 2 + 0] = (float)j1;    // top_idx (as float output)
        out[2 * NO + v * 2 + 1] = (float)j2;
        topidx[v * 2 + 0] = j1;
        topidx[v * 2 + 1] = j2;
    }
}

// ---------------- inversion-bit MLP: one block per (v, j) edge -------------
__global__ __launch_bounds__(128)
void k_inv(const float* __restrict__ h, const float* __restrict__ z,
           const int* __restrict__ topidx,
           const float* __restrict__ w1, const float* __restrict__ b1,
           const float* __restrict__ w2, const float* __restrict__ b2,
           float* __restrict__ out, int n) {
    int p = blockIdx.x;
    int v = p >> 1;
    int u = topidx[p];
    int t = threadIdx.x;

    __shared__ float sin_[3 * HDIM];
    sin_[t]            = h[(size_t)u * HDIM + t];
    sin_[HDIM + t]     = h[(size_t)v * HDIM + t];
    sin_[2 * HDIM + t] = z[t];
    __syncthreads();

    const float* wrow = w1 + (size_t)t * (3 * HDIM);
    float acc = b1[t];
#pragma unroll 8
    for (int k = 0; k < 3 * HDIM; ++k) acc += sin_[k] * wrow[k];
    float part = fmaxf(acc, 0.0f) * w2[t];

    __shared__ float red[128];
    red[t] = part;
    __syncthreads();
    for (int s = 64; s > 0; s >>= 1) {
        if (t < s) red[t] += red[t + s];
        __syncthreads();
    }
    if (t == 0) {
        float logit = red[0] + b2[0];
        int NO = n * 2;
        out[NO + p]     = logit;                          // inv_logit
        out[3 * NO + p] = (logit > 0.0f) ? 1.0f : 0.0f;   // inv_bit
    }
}

__global__ void k_valid(const float* x, const int* depth, const int* lim,
                        float* out, int n) {
    int v = blockIdx.x * blockDim.x + threadIdx.x;
    if (v >= n) return;
    int ty = (int)x[v * 2];
    int tv = (depth[v] >= 1) && (ty != 0);
    int nc = lim[v];
    int NO = n * 2;
    out[4 * NO + v * 2 + 0] = (tv && nc >= 1) ? 1.0f : 0.0f;
    out[4 * NO + v * 2 + 1] = (tv && ty == 2 && nc >= 2) ? 1.0f : 0.0f;
}

// ---------------------------------------------------------------------------
extern "C" void kernel_launch(void* const* d_in, const int* in_sizes, int n_in,
                              void* d_out, int out_size, void* d_ws,
                              size_t ws_size, hipStream_t stream) {
    (void)in_sizes; (void)n_in; (void)out_size; (void)ws_size;
    const int N = NNODES, E = NEDGES;

    const float* x     = (const float*)d_in[0];
    const float* z     = (const float*)d_in[1];
    const int*   eidx  = (const int*)d_in[2];
    const int*   depth = (const int*)d_in[3];
    const float* c1w   = (const float*)d_in[4];
    const float* c1b   = (const float*)d_in[5];
    const float* c2w   = (const float*)d_in[6];
    const float* c2b   = (const float*)d_in[7];
    const float* np1w  = (const float*)d_in[8];
    const float* np1b  = (const float*)d_in[9];
    const float* np2w  = (const float*)d_in[10];
    const float* np2b  = (const float*)d_in[11];
    const float* srcw  = (const float*)d_in[12];
    const float* tgtw  = (const float*)d_in[13];
    const float* i1w   = (const float*)d_in[14];
    const float* i1b   = (const float*)d_in[15];
    const float* i2w   = (const float*)d_in[16];
    const float* i2b   = (const float*)d_in[17];
    const int* esrc = eidx;
    const int* edst = eidx + E;
    float* out = (float*)d_out;

    // ---- workspace layout (all offsets 256B aligned) ----
    char* w = (char*)d_ws;
    size_t o = 0;
    auto take = [&](size_t bytes) { char* p = w + o; o += (bytes + 255) & ~(size_t)255; return p; };
    float*    deg    = (float*)take((size_t)N * 4);
    float*    rsq    = (float*)take((size_t)N * 4);
    int*      lim    = (int*)take((size_t)N * 4);
    int*      topidx = (int*)take((size_t)N * 2 * 4);
    float*    bufA   = (float*)take((size_t)N * HDIM * 4);   // xw / t1 / Ht / Hs staging
    float*    bufB   = (float*)take((size_t)N * HDIM * 4);   // agg
    float*    bufC   = (float*)take((size_t)N * HDIM * 4);   // h1 then h2
    float*    bufD   = (float*)take((size_t)N * HDIM * 4);   // final h (persists)
    _Float16* s16a   = (_Float16*)take((size_t)N * HDIM * 2); // h1_16 / t1_16 / h16
    _Float16* hz16   = (_Float16*)take((size_t)N * 256 * 2);
    _Float16* ht16   = (_Float16*)take((size_t)N * HDIM * 2);
    _Float16* hs16   = (_Float16*)take((size_t)N * HDIM * 2);
    _Float16* w16    = (_Float16*)take((size_t)HDIM * 256 * 2); // shared weight buf

    const int TB = 256;
    const int gN   = (N + TB - 1) / TB;
    const int gE   = (E + TB - 1) / TB;
    const int gNH  = (N * HDIM + TB - 1) / TB;
    const int gEH  = (E * HDIM + TB - 1) / TB;
    const int gNZ  = (N * 256 + TB - 1) / TB;
    const int gemm_blocks = N / 16;   // 512

    // ---- degree / rsqrt / candidate-limit prep ----
    hipMemsetAsync(deg, 0, (size_t)N * 4, stream);
    k_fill_deg<<<gE, TB, 0, stream>>>(edst, deg, E);
    k_node_prep<<<gN, TB, 0, stream>>>(depth, deg, rsq, lim, N);

    // ---- GCN conv1 ----
    k_conv1<<<gNH, TB, 0, stream>>>(x, c1w, bufA, N);
    hipMemsetAsync(bufB, 0, (size_t)N * HDIM * 4, stream);
    k_scatter<<<gEH, TB, 0, stream>>>(esrc, edst, rsq, bufA, bufB, E);
    k_gcn_fin<<<gNH, TB, 0, stream>>>(bufB, bufA, c1b, deg, bufC, N);

    // ---- GCN conv2 (WMMA xw) ----
    k_f32_to_f16<<<gNH, TB, 0, stream>>>(bufC, s16a, N * HDIM);
    k_f32_to_f16<<<(HDIM * HDIM + TB - 1) / TB, TB, 0, stream>>>(c2w, w16, HDIM * HDIM);
    k_gemm_nt<<<gemm_blocks, TB, 0, stream>>>(s16a, w16, bufA, nullptr, HDIM, 0);
    hipMemsetAsync(bufB, 0, (size_t)N * HDIM * 4, stream);
    k_scatter<<<gEH, TB, 0, stream>>>(esrc, edst, rsq, bufA, bufB, E);
    k_gcn_fin<<<gNH, TB, 0, stream>>>(bufB, bufA, c2b, deg, bufC, N);

    // ---- node_proj: relu([h|z] @ np1^T + b) @ np2^T + b ----
    k_build_hz16<<<gNZ, TB, 0, stream>>>(bufC, z, hz16, N);
    k_f32_to_f16<<<(HDIM * 256 + TB - 1) / TB, TB, 0, stream>>>(np1w, w16, HDIM * 256);
    k_gemm_nt<<<gemm_blocks, TB, 0, stream>>>(hz16, w16, bufA, np1b, 256, 1);
    k_f32_to_f16<<<gNH, TB, 0, stream>>>(bufA, s16a, N * HDIM);
    k_f32_to_f16<<<(HDIM * HDIM + TB - 1) / TB, TB, 0, stream>>>(np2w, w16, HDIM * HDIM);
    k_gemm_nt<<<gemm_blocks, TB, 0, stream>>>(s16a, w16, bufD, np2b, HDIM, 0);

    // ---- Ht / Hs projections ----
    k_f32_to_f16<<<gNH, TB, 0, stream>>>(bufD, s16a, N * HDIM);
    k_f32_to_f16<<<(HDIM * HDIM + TB - 1) / TB, TB, 0, stream>>>(tgtw, w16, HDIM * HDIM);
    k_gemm_nt<<<gemm_blocks, TB, 0, stream>>>(s16a, w16, bufA, nullptr, HDIM, 0);
    k_f32_to_f16<<<gNH, TB, 0, stream>>>(bufA, ht16, N * HDIM);
    k_f32_to_f16<<<(HDIM * HDIM + TB - 1) / TB, TB, 0, stream>>>(srcw, w16, HDIM * HDIM);
    k_gemm_nt<<<gemm_blocks, TB, 0, stream>>>(s16a, w16, bufA, nullptr, HDIM, 0);
    k_f32_to_f16<<<gNH, TB, 0, stream>>>(bufA, hs16, N * HDIM);

    // ---- fused NxN scores + mask + top-2 (never materialized) ----
    k_scores_top2<<<gemm_blocks, TB, 0, stream>>>(ht16, hs16, lim, out, topidx, N);

    // ---- inversion-bit MLP + validity flags ----
    k_inv<<<N * 2, 128, 0, stream>>>(bufD, z, topidx, i1w, i1b, i2w, i2b, out, N);
    k_valid<<<gN, TB, 0, stream>>>(x, depth, lim, out, N);
}
